// OPTBlock_687194768143
// MI455X (gfx1250) — matrix operand order repaired and checked by reading, not compiled, not run on hardware
//
#include <hip/hip_runtime.h>
#include <math.h>

// ---------------------------------------------------------------------------
// CDNA5 / gfx1250 windowed-attention transformer block.
// GEMMs on v_wmma_f32_16x16x32_f16; A-tiles staged by the Tensor Data Mover
// (tensor_load_to_lds + s_wait_tensorcnt); everything 128-bit vectorized.
// ---------------------------------------------------------------------------

typedef __attribute__((ext_vector_type(16))) _Float16 v16h;
typedef __attribute__((ext_vector_type(8)))  float    v8f;
typedef __attribute__((ext_vector_type(4)))  unsigned int u32x4;
typedef __attribute__((ext_vector_type(8)))  int          i32x8;
typedef __attribute__((ext_vector_type(4)))  int          i32x4;

#define C_DIM 256
#define NHEAD 8
#define DHEAD 32
#define WINSZ 512
#define HIDD  512
#define EPSV  1e-5f

static __device__ __forceinline__ v8f wmma_f16(v16h a, v16h b, v8f c) {
  return __builtin_amdgcn_wmma_f32_16x16x32_f16(
      false, a, false, b, (short)0, c, false, false);
}

// ---------------------------------------------------------------------------
// TDM: DMA a 2-D f16 tile (tile_rows x 32 elems) from global into LDS with
// 16B row padding (64B data + 16B pad -> ld = 40 halves). D# per ISA ch.8.
// Issue from ONE wave only; follow with s_wait_tensorcnt before the barrier.
// ---------------------------------------------------------------------------
static __device__ __forceinline__ void tdm_load_tile_f16(
    unsigned lds_byte_off, const void* gaddr, unsigned tile_rows,
    unsigned row_stride_elems) {
  unsigned long long ga = (unsigned long long)(uintptr_t)gaddr;
  u32x4 g0;
  g0[0] = 1u;                                    // count=1 (valid user D#)
  g0[1] = lds_byte_off;                          // lds_addr [63:32]
  g0[2] = (unsigned)(ga & 0xFFFFFFFFu);          // global_addr lo
  g0[3] = (unsigned)((ga >> 32) & 0x01FFFFFFu)   // global_addr hi (57b)
          | (2u << 30);                          // type=2 ("image")
  i32x8 g1;
  // word0: data_size=1 (2B)@16 | pad_enable@20 | pad_interval=3 (16 dw)@22
  //        | pad_amount=3 (4 dw)@25 ; workgroup_mask=0 (not in cluster)
  g1[0] = (1 << 16) | (1 << 20) | (3 << 22) | (3 << 25);
  g1[1] = (int)(32u << 16);                      // tensor_dim0 = 32 [79:48] lo
  g1[2] = (int)((tile_rows & 0xFFFFu) << 16);    // dim0 hi | tensor_dim1 lo
  g1[3] = (int)(((tile_rows >> 16) & 0xFFFFu) | (32u << 16)); // dim1 hi | tile_dim0=32
  g1[4] = (int)(tile_rows & 0xFFFFu);            // tile_dim1 | tile_dim2=0
  g1[5] = (int)row_stride_elems;                 // tensor_dim0_stride lo32
  g1[6] = 0;                                     // stride0 hi | stride1 lo
  g1[7] = 0;                                     // stride1 hi
  i32x4 z4 = {0, 0, 0, 0};
#if defined(__clang_major__) && __clang_major__ >= 23
  i32x8 z8 = {0, 0, 0, 0, 0, 0, 0, 0};
  __builtin_amdgcn_tensor_load_to_lds(g0, g1, z4, z4, z8, 0);
#else
  __builtin_amdgcn_tensor_load_to_lds(g0, g1, z4, z4, 0);
#endif
}

// A fragment: 16x32 (MxK) f16, row-major LDS, leading dim ld halves (ISA 7.12.2).
static __device__ __forceinline__ v16h load_a(const _Float16* p, int ld) {
  int lane = threadIdx.x & 31;
  int m = lane & 15;
  int kb = (lane >> 4) << 3;
  const _Float16* q = p + m * ld + kb;
  v16h a;
#pragma unroll
  for (int j = 0; j < 8; ++j) a[j] = q[j];
#pragma unroll
  for (int j = 0; j < 8; ++j) a[8 + j] = q[16 + j];
  return a;
}

// B fragment: 32x16 (KxN), K-major memory: [k][n] at p[k*ld + n].
static __device__ __forceinline__ v16h load_b_kn(const _Float16* p, int ld) {
  int lane = threadIdx.x & 31;
  int n = lane & 15;
  int kb = (lane >> 4) << 3;
  v16h b;
#pragma unroll
  for (int j = 0; j < 16; ++j) b[j] = p[(kb + j + (j >= 8 ? 8 : 0)) * ld + n];
  return b;
}

// B fragment: 32x16 (KxN), N-major memory (W^T): [k][n] at p[n*ld + k].
static __device__ __forceinline__ v16h load_b_nk(const _Float16* p, int ld) {
  int lane = threadIdx.x & 31;
  int n = lane & 15;
  int kb = (lane >> 4) << 3;
  const _Float16* q = p + n * ld + kb;
  v16h b;
#pragma unroll
  for (int j = 0; j < 8; ++j) b[j] = q[j];
#pragma unroll
  for (int j = 0; j < 8; ++j) b[8 + j] = q[16 + j];
  return b;
}

// Stage one 32-col f32 weight tile row-chunk as f16 (vectorized, 8 cols/thread).
static __device__ __forceinline__ void stage_w_f32(
    _Float16* Bs, const float* W, int rows, int c0, int K, int k0) {
  int tid = threadIdx.x;
  for (int i = tid; i < rows * 4; i += 256) {
    int r = i >> 2, cb = (i & 3) * 8;
    const float4 f0 = *(const float4*)&W[(size_t)(c0 + r) * K + k0 + cb];
    const float4 f1 = *(const float4*)&W[(size_t)(c0 + r) * K + k0 + cb + 4];
    union { _Float16 h[8]; uint4 u; } t;
    t.h[0] = (_Float16)f0.x; t.h[1] = (_Float16)f0.y;
    t.h[2] = (_Float16)f0.z; t.h[3] = (_Float16)f0.w;
    t.h[4] = (_Float16)f1.x; t.h[5] = (_Float16)f1.y;
    t.h[6] = (_Float16)f1.z; t.h[7] = (_Float16)f1.w;
    *(uint4*)&Bs[r * 40 + cb] = t.u;
  }
}

// ---------------------------------------------------------------------------
// Kernel 1: LayerNorm1 -> f16; zero scatter accumulators.
// ---------------------------------------------------------------------------
__global__ __launch_bounds__(256)
void k_ln1(const float* __restrict__ x, const float* __restrict__ w,
           const float* __restrict__ b, _Float16* __restrict__ xh,
           float* __restrict__ y, float* __restrict__ counts) {
  __shared__ float red[256];
  int row = blockIdx.x, c = threadIdx.x;
  float v = x[(size_t)row * C_DIM + c];
  red[c] = v; __syncthreads();
  for (int s = 128; s > 0; s >>= 1) { if (c < s) red[c] += red[c + s]; __syncthreads(); }
  float mu = red[0] * (1.f / C_DIM); __syncthreads();
  float d = v - mu;
  red[c] = d * d; __syncthreads();
  for (int s = 128; s > 0; s >>= 1) { if (c < s) red[c] += red[c + s]; __syncthreads(); }
  float var = red[0] * (1.f / C_DIM);
  float o = d * rsqrtf(var + EPSV) * w[c] + b[c];
  xh[(size_t)row * C_DIM + c] = (_Float16)o;
  y[(size_t)row * C_DIM + c] = 0.f;
  if (c == 0) counts[row] = 0.f;
}

// ---------------------------------------------------------------------------
// Kernel 2: QKV GEMM qkv[N,768] = xh[N,256] @ w_qkv[768,256]^T  (f16 out)
// A tile staged by TDM; 128x64 block, 8 waves (4x2), 32x32/wave, K-step 32.
// ---------------------------------------------------------------------------
__global__ __launch_bounds__(256)
void k_gemm_qkv(const _Float16* __restrict__ A, const float* __restrict__ W,
                _Float16* __restrict__ Out) {
  const int K = C_DIM, NOUT = 3 * C_DIM;
  __shared__ _Float16 As[128 * 40];
  __shared__ _Float16 Bs[64 * 40];
  int tid = threadIdx.x;
  int r0 = blockIdx.x * 128, c0 = blockIdx.y * 64;
  int wv = tid >> 5, wm = wv & 3, wn = wv >> 2;
  v8f acc[2][2] = {};
  for (int k0 = 0; k0 < K; k0 += 32) {
    if (tid < 32)
      tdm_load_tile_f16((unsigned)(uintptr_t)As, &A[(size_t)r0 * K + k0], 128, K);
    stage_w_f32(Bs, W, 64, c0, K, k0);
    if (k0 + 32 < K) __builtin_prefetch(&A[(size_t)(r0 + (tid >> 1)) * K + k0 + 32], 0, 0);
    if (tid < 32) __builtin_amdgcn_s_wait_tensorcnt((short)0);
    __syncthreads();
    v16h a0 = load_a(&As[(wm * 32) * 40], 40);
    v16h a1 = load_a(&As[(wm * 32 + 16) * 40], 40);
    v16h b0 = load_b_nk(&Bs[(wn * 32) * 40], 40);
    v16h b1 = load_b_nk(&Bs[(wn * 32 + 16) * 40], 40);
    acc[0][0] = wmma_f16(a0, b0, acc[0][0]);
    acc[0][1] = wmma_f16(a0, b1, acc[0][1]);
    acc[1][0] = wmma_f16(a1, b0, acc[1][0]);
    acc[1][1] = wmma_f16(a1, b1, acc[1][1]);
    __syncthreads();
  }
  int lane = tid & 31, n = lane & 15, hig = lane >> 4;
#pragma unroll
  for (int fm = 0; fm < 2; ++fm)
#pragma unroll
    for (int fn = 0; fn < 2; ++fn)
#pragma unroll
      for (int r = 0; r < 8; ++r) {
        int row = r0 + wm * 32 + fm * 16 + r + hig * 8;
        int col = c0 + wn * 32 + fn * 16 + n;
        Out[(size_t)row * NOUT + col] = (_Float16)acc[fm][fn][r];
      }
}

// ---------------------------------------------------------------------------
// Kernel 3: windowed attention, one block per (window, head).
// ---------------------------------------------------------------------------
__global__ __launch_bounds__(256)
void k_attn(const _Float16* __restrict__ qkv, const int* __restrict__ win_idx,
            const unsigned char* __restrict__ win_mask,
            _Float16* __restrict__ attn_out) {
  __shared__ _Float16 Ks[WINSZ * 40];
  __shared__ _Float16 Vs[WINSZ * 40];
  __shared__ float    msk[WINSZ];
  __shared__ _Float16 Ps[8 * 16 * 40];
  int win = blockIdx.x >> 3;
  int h   = blockIdx.x & 7;
  int tid = threadIdx.x;
  for (int i = tid; i < WINSZ; i += 256) {
    int t = win_idx[win * WINSZ + i];
    const uint4* kp = (const uint4*)(qkv + (size_t)t * (3 * C_DIM) + C_DIM + h * DHEAD);
    const uint4* vp = (const uint4*)(qkv + (size_t)t * (3 * C_DIM) + 2 * C_DIM + h * DHEAD);
    uint4* kd = (uint4*)&Ks[i * 40];
    uint4* vd = (uint4*)&Vs[i * 40];
#pragma unroll
    for (int j = 0; j < 4; ++j) { kd[j] = kp[j]; vd[j] = vp[j]; }
    msk[i] = win_mask[win * WINSZ + i] ? 0.f : -1e30f;
  }
  __syncthreads();
  int lane = tid & 31, wv = tid >> 5;
  int n = lane & 15, hig = lane >> 4;
  const float rscale = 0.17677669529663687f;  // 1/sqrt(32)
  for (int strip = wv; strip < WINSZ / 16; strip += 8) {
    int q0 = strip * 16;
    int t = win_idx[win * WINSZ + q0 + n];
    const char* qp = (const char*)(qkv + (size_t)t * (3 * C_DIM) + h * DHEAD + (hig << 3));
    v16h aQ;
    *(uint4*)&aQ       = *(const uint4*)qp;          // halves 0..7  (K kb..kb+7)
    *((uint4*)&aQ + 1) = *(const uint4*)(qp + 32);   // halves 8..15 (K kb+16..kb+23)
    v8f o0 = {}, o1 = {};
    float mrun[8], lrun[8];
#pragma unroll
    for (int r = 0; r < 8; ++r) { mrun[r] = -1e30f; lrun[r] = 0.f; }
    for (int kt = 0; kt < WINSZ; kt += 32) {
      v16h bk0 = load_b_nk(&Ks[kt * 40], 40);
      v16h bk1 = load_b_nk(&Ks[(kt + 16) * 40], 40);
      v8f s0 = {}, s1 = {};
      s0 = wmma_f16(aQ, bk0, s0);
      s1 = wmma_f16(aQ, bk1, s1);
      float am0 = msk[kt + n], am1 = msk[kt + 16 + n];
      float p0[8], p1[8];
#pragma unroll
      for (int r = 0; r < 8; ++r) {
        float v0 = s0[r] * rscale + am0;
        float v1 = s1[r] * rscale + am1;
        float rm = fmaxf(v0, v1);
        rm = fmaxf(rm, __shfl_xor(rm, 1, 32));
        rm = fmaxf(rm, __shfl_xor(rm, 2, 32));
        rm = fmaxf(rm, __shfl_xor(rm, 4, 32));
        rm = fmaxf(rm, __shfl_xor(rm, 8, 32));
        float mnew = fmaxf(mrun[r], rm);
        float alpha = __expf(mrun[r] - mnew);
        mrun[r] = mnew;
        p0[r] = __expf(v0 - mnew);
        p1[r] = __expf(v1 - mnew);
        float rs = p0[r] + p1[r];
        rs += __shfl_xor(rs, 1, 32);
        rs += __shfl_xor(rs, 2, 32);
        rs += __shfl_xor(rs, 4, 32);
        rs += __shfl_xor(rs, 8, 32);
        lrun[r] = lrun[r] * alpha + rs;
        o0[r] *= alpha; o1[r] *= alpha;
      }
      _Float16* pw = &Ps[wv * 16 * 40];
#pragma unroll
      for (int r = 0; r < 8; ++r) {
        pw[(r + hig * 8) * 40 + n]      = (_Float16)p0[r];
        pw[(r + hig * 8) * 40 + 16 + n] = (_Float16)p1[r];
      }
      asm volatile("s_wait_dscnt 0" ::: "memory");  // same-wave LDS W->R
      v16h aP  = load_a(pw, 40);
      v16h bv0 = load_b_kn(&Vs[kt * 40], 40);
      v16h bv1 = load_b_kn(&Vs[kt * 40 + 16], 40);
      o0 = wmma_f16(aP, bv0, o0);
      o1 = wmma_f16(aP, bv1, o1);
    }
#pragma unroll
    for (int r = 0; r < 8; ++r) {
      float inv = 1.f / lrun[r];
      int row = win * WINSZ + q0 + r + hig * 8;
      size_t base = (size_t)row * C_DIM + h * DHEAD;
      attn_out[base + n]      = (_Float16)(o0[r] * inv);
      attn_out[base + 16 + n] = (_Float16)(o1[r] * inv);
    }
  }
}

// ---------------------------------------------------------------------------
// Kernel 4a: per-token overlap counts.
// ---------------------------------------------------------------------------
__global__ __launch_bounds__(256)
void k_counts(const int* __restrict__ win_idx, const unsigned char* __restrict__ mask,
              float* __restrict__ counts, int mw) {
  int i = blockIdx.x * 256 + threadIdx.x;
  if (i < mw && mask[i]) atomicAdd(&counts[win_idx[i]], 1.f);
}

// ---------------------------------------------------------------------------
// Kernel 4b: out-projection GEMM + masked atomic scatter-add into y.
// ---------------------------------------------------------------------------
__global__ __launch_bounds__(256)
void k_proj_scatter(const _Float16* __restrict__ A, const float* __restrict__ W,
                    const float* __restrict__ bias, const int* __restrict__ win_idx,
                    const unsigned char* __restrict__ mask, float* __restrict__ y) {
  const int K = C_DIM;
  __shared__ _Float16 As[128 * 40];
  __shared__ _Float16 Bs[64 * 40];
  int tid = threadIdx.x;
  int r0 = blockIdx.x * 128, c0 = blockIdx.y * 64;
  int wv = tid >> 5, wm = wv & 3, wn = wv >> 2;
  v8f acc[2][2] = {};
  for (int k0 = 0; k0 < K; k0 += 32) {
    if (tid < 32)
      tdm_load_tile_f16((unsigned)(uintptr_t)As, &A[(size_t)r0 * K + k0], 128, K);
    stage_w_f32(Bs, W, 64, c0, K, k0);
    if (k0 + 32 < K) __builtin_prefetch(&A[(size_t)(r0 + (tid >> 1)) * K + k0 + 32], 0, 0);
    if (tid < 32) __builtin_amdgcn_s_wait_tensorcnt((short)0);
    __syncthreads();
    v16h a0 = load_a(&As[(wm * 32) * 40], 40);
    v16h a1 = load_a(&As[(wm * 32 + 16) * 40], 40);
    v16h b0 = load_b_nk(&Bs[(wn * 32) * 40], 40);
    v16h b1 = load_b_nk(&Bs[(wn * 32 + 16) * 40], 40);
    acc[0][0] = wmma_f16(a0, b0, acc[0][0]);
    acc[0][1] = wmma_f16(a0, b1, acc[0][1]);
    acc[1][0] = wmma_f16(a1, b0, acc[1][0]);
    acc[1][1] = wmma_f16(a1, b1, acc[1][1]);
    __syncthreads();
  }
  int lane = tid & 31, n = lane & 15, hig = lane >> 4;
#pragma unroll
  for (int fm = 0; fm < 2; ++fm)
#pragma unroll
    for (int fn = 0; fn < 2; ++fn)
#pragma unroll
      for (int r = 0; r < 8; ++r) {
        int row = r0 + wm * 32 + fm * 16 + r + hig * 8;
        int col = c0 + wn * 32 + fn * 16 + n;
        if (mask[row]) {
          float v = acc[fm][fn][r] + bias[col];
          atomicAdd(&y[(size_t)win_idx[row] * C_DIM + col], v);
        }
      }
}

// ---------------------------------------------------------------------------
// Kernel 5: x1 = x + y/max(counts,1); LN2(x1) -> f16.
// ---------------------------------------------------------------------------
__global__ __launch_bounds__(256)
void k_x1_ln2(const float* __restrict__ x, const float* __restrict__ y,
              const float* __restrict__ counts, const float* __restrict__ w,
              const float* __restrict__ b, float* __restrict__ x1,
              _Float16* __restrict__ x2h) {
  __shared__ float red[256];
  int row = blockIdx.x, c = threadIdx.x;
  float cnt = fmaxf(counts[row], 1.f);
  float v = x[(size_t)row * C_DIM + c] + y[(size_t)row * C_DIM + c] / cnt;
  x1[(size_t)row * C_DIM + c] = v;
  red[c] = v; __syncthreads();
  for (int s = 128; s > 0; s >>= 1) { if (c < s) red[c] += red[c + s]; __syncthreads(); }
  float mu = red[0] * (1.f / C_DIM); __syncthreads();
  float d = v - mu;
  red[c] = d * d; __syncthreads();
  for (int s = 128; s > 0; s >>= 1) { if (c < s) red[c] += red[c + s]; __syncthreads(); }
  float var = red[0] * (1.f / C_DIM);
  x2h[(size_t)row * C_DIM + c] = (_Float16)(d * rsqrtf(var + EPSV) * w[c] + b[c]);
}

// ---------------------------------------------------------------------------
// Kernel 6: FFN1 GEMM h[N,512] = x2h[N,256] @ w1[512,256]^T + b1  (f32 out)
// ---------------------------------------------------------------------------
__global__ __launch_bounds__(256)
void k_ffn1(const _Float16* __restrict__ A, const float* __restrict__ W,
            const float* __restrict__ bias, float* __restrict__ Out) {
  const int K = C_DIM, NOUT = HIDD;
  __shared__ _Float16 As[128 * 40];
  __shared__ _Float16 Bs[64 * 40];
  int tid = threadIdx.x;
  int r0 = blockIdx.x * 128, c0 = blockIdx.y * 64;
  int wv = tid >> 5, wm = wv & 3, wn = wv >> 2;
  v8f acc[2][2] = {};
  for (int k0 = 0; k0 < K; k0 += 32) {
    if (tid < 32)
      tdm_load_tile_f16((unsigned)(uintptr_t)As, &A[(size_t)r0 * K + k0], 128, K);
    stage_w_f32(Bs, W, 64, c0, K, k0);
    if (k0 + 32 < K) __builtin_prefetch(&A[(size_t)(r0 + (tid >> 1)) * K + k0 + 32], 0, 0);
    if (tid < 32) __builtin_amdgcn_s_wait_tensorcnt((short)0);
    __syncthreads();
    v16h a0 = load_a(&As[(wm * 32) * 40], 40);
    v16h a1 = load_a(&As[(wm * 32 + 16) * 40], 40);
    v16h b0 = load_b_nk(&Bs[(wn * 32) * 40], 40);
    v16h b1 = load_b_nk(&Bs[(wn * 32 + 16) * 40], 40);
    acc[0][0] = wmma_f16(a0, b0, acc[0][0]);
    acc[0][1] = wmma_f16(a0, b1, acc[0][1]);
    acc[1][0] = wmma_f16(a1, b0, acc[1][0]);
    acc[1][1] = wmma_f16(a1, b1, acc[1][1]);
    __syncthreads();
  }
  int lane = tid & 31, n = lane & 15, hig = lane >> 4;
#pragma unroll
  for (int fm = 0; fm < 2; ++fm)
#pragma unroll
    for (int fn = 0; fn < 2; ++fn)
#pragma unroll
      for (int r = 0; r < 8; ++r) {
        int row = r0 + wm * 32 + fm * 16 + r + hig * 8;
        int col = c0 + wn * 32 + fn * 16 + n;
        Out[(size_t)row * NOUT + col] = acc[fm][fn][r] + bias[col];
      }
}

// ---------------------------------------------------------------------------
// Kernel 7: BatchNorm statistics (training mode): mu[c], istd[c] over batch.
// ---------------------------------------------------------------------------
__global__ __launch_bounds__(256)
void k_bn_stats(const float* __restrict__ h, float* __restrict__ mu,
                float* __restrict__ istd, int nrows) {
  __shared__ float red[256];
  int c = blockIdx.x, tid = threadIdx.x;
  float s = 0.f, s2 = 0.f;
  for (int r = tid; r < nrows; r += 256) {
    float v = h[(size_t)r * HIDD + c];
    s += v; s2 += v * v;
  }
  red[tid] = s; __syncthreads();
  for (int k = 128; k > 0; k >>= 1) { if (tid < k) red[tid] += red[tid + k]; __syncthreads(); }
  float m = red[0] / (float)nrows; __syncthreads();
  red[tid] = s2; __syncthreads();
  for (int k = 128; k > 0; k >>= 1) { if (tid < k) red[tid] += red[tid + k]; __syncthreads(); }
  float var = red[0] / (float)nrows - m * m;
  if (tid == 0) { mu[c] = m; istd[c] = rsqrtf(var + EPSV); }
}

// ---------------------------------------------------------------------------
// Kernel 8: FFN2 GEMM with fused BN+ReLU A-staging and residual epilogue.
// ---------------------------------------------------------------------------
__global__ __launch_bounds__(256)
void k_ffn2(const float* __restrict__ Hb, const float* __restrict__ mu,
            const float* __restrict__ istd, const float* __restrict__ bnw,
            const float* __restrict__ bnb, const float* __restrict__ W,
            const float* __restrict__ bias, const float* __restrict__ x1,
            float* __restrict__ Out) {
  const int K = HIDD, NOUT = C_DIM;
  __shared__ _Float16 As[128 * 40];
  __shared__ _Float16 Bs[64 * 40];
  int tid = threadIdx.x;
  int r0 = blockIdx.x * 128, c0 = blockIdx.y * 64;
  int wv = tid >> 5, wm = wv & 3, wn = wv >> 2;
  v8f acc[2][2] = {};
  for (int k0 = 0; k0 < K; k0 += 32) {
    for (int i = tid; i < 128 * 4; i += 256) {
      int r = i >> 2, cb = (i & 3) * 8;
      int k = k0 + cb;
      const float4 f0 = *(const float4*)&Hb[(size_t)(r0 + r) * K + k];
      const float4 f1 = *(const float4*)&Hb[(size_t)(r0 + r) * K + k + 4];
      union { _Float16 h[8]; uint4 u; } t;
      float vv[8] = {f0.x, f0.y, f0.z, f0.w, f1.x, f1.y, f1.z, f1.w};
#pragma unroll
      for (int j = 0; j < 8; ++j) {
        float v = (vv[j] - mu[k + j]) * istd[k + j] * bnw[k + j] + bnb[k + j];
        t.h[j] = (_Float16)fmaxf(v, 0.f);
      }
      *(uint4*)&As[r * 40 + cb] = t.u;
    }
    stage_w_f32(Bs, W, 64, c0, K, k0);
    if (k0 + 32 < K) __builtin_prefetch(&Hb[(size_t)(r0 + (tid >> 1)) * K + k0 + 32], 0, 0);
    __syncthreads();
    v16h a0 = load_a(&As[(wm * 32) * 40], 40);
    v16h a1 = load_a(&As[(wm * 32 + 16) * 40], 40);
    v16h b0 = load_b_nk(&Bs[(wn * 32) * 40], 40);
    v16h b1 = load_b_nk(&Bs[(wn * 32 + 16) * 40], 40);
    acc[0][0] = wmma_f16(a0, b0, acc[0][0]);
    acc[0][1] = wmma_f16(a0, b1, acc[0][1]);
    acc[1][0] = wmma_f16(a1, b0, acc[1][0]);
    acc[1][1] = wmma_f16(a1, b1, acc[1][1]);
    __syncthreads();
  }
  int lane = tid & 31, n = lane & 15, hig = lane >> 4;
#pragma unroll
  for (int fm = 0; fm < 2; ++fm)
#pragma unroll
    for (int fn = 0; fn < 2; ++fn)
#pragma unroll
      for (int r = 0; r < 8; ++r) {
        int row = r0 + wm * 32 + fm * 16 + r + hig * 8;
        int col = c0 + wn * 32 + fn * 16 + n;
        Out[(size_t)row * NOUT + col] =
            x1[(size_t)row * NOUT + col] + acc[fm][fn][r] + bias[col];
      }
}

// ---------------------------------------------------------------------------
// Host launcher
// ---------------------------------------------------------------------------
extern "C" void kernel_launch(void* const* d_in, const int* in_sizes, int n_in,
                              void* d_out, int out_size, void* d_ws, size_t ws_size,
                              hipStream_t stream) {
  const float*         x        = (const float*)d_in[0];
  const int*           win_idx  = (const int*)d_in[1];
  const unsigned char* win_mask = (const unsigned char*)d_in[2];  // jax bool = u8
  const float* ln1_w = (const float*)d_in[3];
  const float* ln1_b = (const float*)d_in[4];
  const float* w_qkv = (const float*)d_in[5];
  const float* w_proj= (const float*)d_in[6];
  const float* b_proj= (const float*)d_in[7];
  const float* ln2_w = (const float*)d_in[8];
  const float* ln2_b = (const float*)d_in[9];
  const float* w1    = (const float*)d_in[10];
  const float* b1    = (const float*)d_in[11];
  const float* bn_w  = (const float*)d_in[12];
  const float* bn_b  = (const float*)d_in[13];
  const float* w2    = (const float*)d_in[14];
  const float* b2    = (const float*)d_in[15];

  int N  = in_sizes[0] / C_DIM;   // 25600 tokens
  int MW = in_sizes[1];           // M * WIN = 51200
  int M  = MW / WINSZ;            // 100 windows

  char* p = (char*)d_ws;
  auto carve = [&](size_t bytes) { char* r = p; p += (bytes + 255) & ~(size_t)255; return (void*)r; };
  _Float16* xh     = (_Float16*)carve((size_t)N * C_DIM * 2);     // dead after qkv GEMM
  _Float16* qkvh   = (_Float16*)carve((size_t)N * 3 * C_DIM * 2); // dead after attn
  _Float16* attn   = (_Float16*)carve((size_t)MW * C_DIM * 2);    // dead after proj
  float*    ybuf   = (float*)carve((size_t)N * C_DIM * 4);
  float*    counts = (float*)carve((size_t)N * 4);
  float*    x1     = (float*)carve((size_t)N * C_DIM * 4);
  float*    mu     = (float*)carve(HIDD * 4);
  float*    istd   = (float*)carve(HIDD * 4);
  _Float16* x2h    = xh;                                          // alias (lifetime-safe)
  float*    hbuf   = (float*)qkvh;                                // alias qkvh+attn region

  k_ln1<<<N, 256, 0, stream>>>(x, ln1_w, ln1_b, xh, ybuf, counts);
  k_gemm_qkv<<<dim3(N / 128, (3 * C_DIM) / 64), 256, 0, stream>>>(xh, w_qkv, qkvh);
  k_attn<<<M * NHEAD, 256, 0, stream>>>(qkvh, win_idx, win_mask, attn);
  k_counts<<<(MW + 255) / 256, 256, 0, stream>>>(win_idx, win_mask, counts, MW);
  k_proj_scatter<<<dim3(MW / 128, C_DIM / 64), 256, 0, stream>>>(
      attn, w_proj, b_proj, win_idx, win_mask, ybuf);
  k_x1_ln2<<<N, 256, 0, stream>>>(x, ybuf, counts, ln2_w, ln2_b, x1, x2h);
  k_ffn1<<<dim3(N / 128, HIDD / 64), 256, 0, stream>>>(x2h, w1, b1, hbuf);
  k_bn_stats<<<HIDD, 256, 0, stream>>>(hbuf, mu, istd, N);
  k_ffn2<<<dim3(N / 128, C_DIM / 64), 256, 0, stream>>>(
      hbuf, mu, istd, bn_w, bn_b, w2, b2, x1, (float*)d_out);
}
